// SpikingNeuralNetwork_42133629174356
// MI455X (gfx1250) — matrix operand order repaired and checked by reading, not compiled, not run on hardware
//
#include <hip/hip_runtime.h>

typedef __attribute__((ext_vector_type(16))) _Float16 v16h;
typedef __attribute__((ext_vector_type(8)))  _Float16 v8h;
typedef __attribute__((ext_vector_type(8)))  float    v8f;

#define SNN_BETA   0.9f
#define SNN_VTH    1.0f
#define SNN_STEPS  16
#define SNN_B      512
#define SNN_T      50
#define SNN_F      2048
#define SNN_H1     1024
#define SNN_H2     1024

// ---------------------------------------------------------------------------
// WMMA fragment layouts (CDNA5 ISA 7.12.2, wave32).
// A (16x32 f16, MxK): lane L holds row m = L&15.
//   VGPR0..3: k = (L<16 ? 0..7  : 8..15), VGPR4..7: k = (L<16 ? 16..23 : 24..31)
// B (32x16 f16, KxN): lane L holds col n = L&15.
//   same k order as A: VGPR0..3: k = (L<16 ? 0..7 : 8..15), VGPR4..7: +16
// C/D (16x16 f32): n = L&15, VGPR r -> m = r + (L>=16 ? 8 : 0).
// ---------------------------------------------------------------------------
__device__ __forceinline__ v16h make_frag(v8h lo, v8h hi) {
    return __builtin_shufflevector(lo, hi, 0,1,2,3,4,5,6,7,8,9,10,11,12,13,14,15);
}

__device__ __forceinline__ v16h load_a_frag(const _Float16* __restrict__ row,
                                            int kb, int lane) {
    const int off = (lane >> 4) << 3;                 // 0 or 8
    v8h lo = *(const v8h*)(row + kb + off);
    v8h hi = *(const v8h*)(row + kb + 16 + off);
    return make_frag(lo, hi);
}

__device__ __forceinline__ v16h load_b_frag(const _Float16* __restrict__ row,
                                            int kb, int lane) {
    const int off = (lane >> 4) << 3;                 // 0 or 8
    v8h lo = *(const v8h*)(row + kb + off);
    v8h hi = *(const v8h*)(row + kb + 16 + off);
    return make_frag(lo, hi);
}

// ---------------------------------------------------------------------------
// CDNA5 async global->LDS copy (ASYNCcnt-tracked, ISA 10.7 / 15.18.3).
// Each active lane copies 16B from its own global address to its own LDS
// offset. Flat LDS pointers carry the logical LDS byte offset in addr[31:0].
// ---------------------------------------------------------------------------
__device__ __forceinline__ void async_copy_b128(void* lds_ptr, const void* gptr) {
    unsigned lds_off = (unsigned)(unsigned long long)lds_ptr;
    asm volatile("global_load_async_to_lds_b128 %0, %1, off"
                 :: "v"(lds_off), "v"(gptr)
                 : "memory");
}

__device__ __forceinline__ void wait_asynccnt_8() {
    asm volatile("s_wait_asynccnt 0x8" ::: "memory");
}
__device__ __forceinline__ void wait_asynccnt_0() {
    asm volatile("s_wait_asynccnt 0x0" ::: "memory");
}

// ---------------------------------------------------------------------------
// Kernel 1: rate encoder. count[b,f] = #{t : x[b,t,f] > 0}, stored as f16.
// (sigmoid(x) > 0.5  <=>  x > 0). 210MB streamed: this is the HBM roofline.
// ---------------------------------------------------------------------------
__global__ void __launch_bounds__(256) snn_encode(const float* __restrict__ x,
                                                  _Float16* __restrict__ cnt) {
    const int b  = blockIdx.x;                 // 512
    const int f0 = threadIdx.x << 3;           // 256 threads * 8 = 2048
    int c[8] = {0, 0, 0, 0, 0, 0, 0, 0};
    const float4* base4 = (const float4*)(x + (size_t)b * SNN_T * SNN_F);
    for (int t = 0; t < SNN_T; ++t) {
        const float4* p = base4 + ((t * SNN_F + f0) >> 2);
        float4 a = p[0];
        float4 d = p[1];
        c[0] += (a.x > 0.0f); c[1] += (a.y > 0.0f);
        c[2] += (a.z > 0.0f); c[3] += (a.w > 0.0f);
        c[4] += (d.x > 0.0f); c[5] += (d.y > 0.0f);
        c[6] += (d.z > 0.0f); c[7] += (d.w > 0.0f);
    }
    v8h o;
    #pragma unroll
    for (int i = 0; i < 8; ++i) o[i] = (_Float16)(float)c[i];
    *(v8h*)(cnt + (size_t)b * SNN_F + f0) = o;
}

// ---------------------------------------------------------------------------
// Kernel 2: split fp32 weights into f16 hi + f16 residual (near-fp32 WMMA).
// ---------------------------------------------------------------------------
__global__ void snn_split_w(const float* __restrict__ w,
                            _Float16* __restrict__ hi,
                            _Float16* __restrict__ lo, int n) {
    const int i = blockIdx.x * blockDim.x + threadIdx.x;
    if (i < n) {
        const float v = w[i];
        const _Float16 h = (_Float16)v;
        hi[i] = h;
        lo[i] = (_Float16)(v - (float)h);
    }
}

// ---------------------------------------------------------------------------
// Kernel 3: GEMM1 (count @ W1^T, hi+lo split) fused with LIF1.
// Each wave computes 4 M-tiles x 1 N-tile, reusing the (double-width) B
// fragments 4x. Direct global loads + prefetch (pipelines well, see round 1).
// ---------------------------------------------------------------------------
__global__ void __launch_bounds__(256)
snn_gemm1_lif1(const _Float16* __restrict__ cnt,
               const _Float16* __restrict__ w1hi,
               const _Float16* __restrict__ w1lo,
               const float*    __restrict__ b1,
               _Float16*       __restrict__ spk1) {
    const int lane = threadIdx.x & 31;
    const int wv   = threadIdx.x >> 5;
    const int tile = blockIdx.x * 8 + wv;      // 8 mg * 64 nt = 512 wave-tiles
    const int mg = tile >> 6;                  // 0..7, 64 rows each
    const int nt = tile & 63;
    const int b0 = mg << 6;
    const int h0 = nt << 4;

    const _Float16* bhr = w1hi + (size_t)(h0 + (lane & 15)) * SNN_F;
    const _Float16* blr = w1lo + (size_t)(h0 + (lane & 15)) * SNN_F;
    const _Float16* ar[4];
    #pragma unroll
    for (int i = 0; i < 4; ++i)
        ar[i] = cnt + (size_t)(b0 + i * 16 + (lane & 15)) * SNN_F;

    v8f acc[4];
    const v8f zero = {};
    #pragma unroll
    for (int i = 0; i < 4; ++i) acc[i] = zero;

    for (int kb = 0; kb < SNN_F; kb += 32) {
        v16h bh = load_b_frag(bhr, kb, lane);
        v16h bl = load_b_frag(blr, kb, lane);
        __builtin_prefetch(bhr + kb + 32, 0, 3);
        __builtin_prefetch(blr + kb + 32, 0, 3);
        #pragma unroll
        for (int i = 0; i < 4; ++i) {
            v16h a = load_a_frag(ar[i], kb, lane);
            acc[i] = __builtin_amdgcn_wmma_f32_16x16x32_f16(false, a, false, bh,
                                                            (short)0, acc[i],
                                                            false, false);
            acc[i] = __builtin_amdgcn_wmma_f32_16x16x32_f16(false, a, false, bl,
                                                            (short)0, acc[i],
                                                            false, false);
        }
    }

    const int   n     = lane & 15;
    const int   mbase = (lane >> 4) << 3;
    const float bias  = b1[h0 + n];
    #pragma unroll
    for (int i = 0; i < 4; ++i) {
        #pragma unroll
        for (int r = 0; r < 8; ++r) {
            const int   row = b0 + i * 16 + mbase + r;
            const float c   = acc[i][r] * (1.0f / (float)SNN_T) + bias;
            float mem = 0.0f;
            _Float16* outp = spk1 + (size_t)row * SNN_H1 + h0 + n;
            #pragma unroll
            for (int s = 0; s < SNN_STEPS; ++s) {
                const float reset = (mem > SNN_VTH) ? SNN_VTH : 0.0f;
                mem = SNN_BETA * mem + c - reset;
                outp[(size_t)s * SNN_B * SNN_H1] =
                    (_Float16)((mem > SNN_VTH) ? 1.0f : 0.0f);
            }
        }
    }
}

// ---------------------------------------------------------------------------
// Kernel 4: GEMM2 over all 16 steps fused with LIF2, LDS-staged.
// Block = 8 waves = 1 M-tile x 8 N-tiles x 16 steps. Per K-step the block
// async-copies (ASYNCcnt, double-buffered) the shared A slab (16 steps x
// 16x32 spk1) and B slab (128x32 W2 hi+lo) into LDS; each wave then feeds its
// 16 per-step accumulators with ds_load fragments -> 32 WMMAs per K-step.
// ---------------------------------------------------------------------------
__global__ void __launch_bounds__(256)
snn_gemm2_lif2(const _Float16* __restrict__ spk1,
               const _Float16* __restrict__ w2hi,
               const _Float16* __restrict__ w2lo,
               const float*    __restrict__ b2,
               float*          __restrict__ out) {
    const int tid  = threadIdx.x;
    const int lane = tid & 31;
    const int wv   = tid >> 5;                 // 0..7 -> N-tile in group
    const int mt   = blockIdx.x >> 3;          // 0..31
    const int ntg  = blockIdx.x & 7;           // 0..7
    const int b0   = mt << 4;
    const int h0g  = ntg << 7;                 // 128 columns per block
    const int h0   = h0g + (wv << 4);

    __shared__ _Float16 sA [2][SNN_STEPS * 16 * 32];  // 2 x 16KB
    __shared__ _Float16 sBh[2][128 * 32];             // 2 x 8KB
    __shared__ _Float16 sBl[2][128 * 32];             // 2 x 8KB

    // Per-thread async-copy assignments (per K-step stage):
    //   A: thread t -> (s = t>>4, row = t&15), 4 x b128 (64B row slab)
    //   B: thread t<128 -> W2hi row t; t>=128 -> W2lo row t-128; 4 x b128
    const int aS   = tid >> 4;
    const int aRow = tid & 15;
    const _Float16* aG = spk1 + ((size_t)(aS * SNN_B + b0 + aRow)) * SNN_H1;
    const int bRow = tid & 127;
    const _Float16* bG = ((tid < 128) ? w2hi : w2lo) +
                         (size_t)(h0g + bRow) * SNN_H1;
    _Float16* aL0 = &sA[0][(aS * 16 + aRow) * 32];
    _Float16* aL1 = &sA[1][(aS * 16 + aRow) * 32];
    _Float16* bL0 = (tid < 128) ? &sBh[0][bRow * 32] : &sBl[0][bRow * 32];
    _Float16* bL1 = (tid < 128) ? &sBh[1][bRow * 32] : &sBl[1][bRow * 32];

    v8f acc[SNN_STEPS];
    const v8f zero = {};
    #pragma unroll
    for (int s = 0; s < SNN_STEPS; ++s) acc[s] = zero;

    const int nLoc  = (wv << 4) + (lane & 15);        // B row in LDS slab
    const int offB  = (lane >> 4) << 3;               // 0 or 8
    const int offA  = (lane >> 4) << 3;               // 0 or 8
    const int rowA  = lane & 15;

    // Prologue: stage 0 into buffer 0.
    #pragma unroll
    for (int c = 0; c < 4; ++c) {
        async_copy_b128(aL0 + c * 8, aG + c * 8);
        async_copy_b128(bL0 + c * 8, bG + c * 8);
    }

    const int nSteps = SNN_H1 / 32;                   // 32 K-steps
    for (int ki = 0; ki < nSteps; ++ki) {
        const int buf = ki & 1;
        if (ki + 1 < nSteps) {
            const int kb = (ki + 1) * 32;
            _Float16* aL = buf ? aL0 : aL1;
            _Float16* bL = buf ? bL0 : bL1;
            #pragma unroll
            for (int c = 0; c < 4; ++c) {
                async_copy_b128(aL + c * 8, aG + kb + c * 8);
                async_copy_b128(bL + c * 8, bG + kb + c * 8);
            }
            wait_asynccnt_8();    // stage ki complete (async loads retire in order)
        } else {
            wait_asynccnt_0();
        }
        __syncthreads();

        const _Float16* pBh = &sBh[buf][nLoc * 32];
        const _Float16* pBl = &sBl[buf][nLoc * 32];
        v16h bh = make_frag(*(const v8h*)(pBh + offB), *(const v8h*)(pBh + 16 + offB));
        v16h bl = make_frag(*(const v8h*)(pBl + offB), *(const v8h*)(pBl + 16 + offB));
        #pragma unroll
        for (int s = 0; s < SNN_STEPS; ++s) {
            const _Float16* pA = &sA[buf][(s * 16 + rowA) * 32];
            v16h a = make_frag(*(const v8h*)(pA + offA),
                               *(const v8h*)(pA + 16 + offA));
            acc[s] = __builtin_amdgcn_wmma_f32_16x16x32_f16(false, a, false, bh,
                                                            (short)0, acc[s],
                                                            false, false);
            acc[s] = __builtin_amdgcn_wmma_f32_16x16x32_f16(false, a, false, bl,
                                                            (short)0, acc[s],
                                                            false, false);
        }
        __syncthreads();   // protect buffer before next stage overwrites it
    }

    // LIF2 epilogue: recurrence across the 16 per-step accumulators.
    const int   n     = lane & 15;
    const int   mbase = (lane >> 4) << 3;
    const float bias  = b2[h0 + n];
    #pragma unroll
    for (int r = 0; r < 8; ++r) {
        const int row = b0 + mbase + r;
        float mem = 0.0f;
        float* outp = out + (size_t)row * SNN_H2 + h0 + n;
        #pragma unroll
        for (int s = 0; s < SNN_STEPS; ++s) {
            const float c     = acc[s][r] + bias;
            const float reset = (mem > SNN_VTH) ? SNN_VTH : 0.0f;
            mem = SNN_BETA * mem + c - reset;
            outp[(size_t)s * SNN_B * SNN_H2] = (mem > SNN_VTH) ? 1.0f : 0.0f;
        }
    }
}

// ---------------------------------------------------------------------------
// Launch: encoder -> weight splits -> GEMM1+LIF1 -> GEMM2+LIF2 (one stream).
// Workspace layout (bytes):
//   [0,       2MB)  count  f16 [512, 2048]
//   [2MB,     6MB)  W1 hi  f16 [1024, 2048]
//   [6MB,    10MB)  W1 lo  f16 [1024, 2048]
//   [10MB,   12MB)  W2 hi  f16 [1024, 1024]
//   [12MB,   14MB)  W2 lo  f16 [1024, 1024]
//   [14MB,   30MB)  spk1   f16 [16, 512, 1024]
// ---------------------------------------------------------------------------
extern "C" void kernel_launch(void* const* d_in, const int* in_sizes, int n_in,
                              void* d_out, int out_size, void* d_ws, size_t ws_size,
                              hipStream_t stream) {
    (void)in_sizes; (void)n_in; (void)out_size; (void)ws_size;

    const float* x  = (const float*)d_in[0];
    const float* W1 = (const float*)d_in[1];
    const float* b1 = (const float*)d_in[2];
    const float* W2 = (const float*)d_in[3];
    const float* b2 = (const float*)d_in[4];
    float* out = (float*)d_out;

    char* ws = (char*)d_ws;
    const size_t MB = 1024u * 1024u;
    _Float16* cnt  = (_Float16*)(ws + 0 * MB);
    _Float16* w1hi = (_Float16*)(ws + 2 * MB);
    _Float16* w1lo = (_Float16*)(ws + 6 * MB);
    _Float16* w2hi = (_Float16*)(ws + 10 * MB);
    _Float16* w2lo = (_Float16*)(ws + 12 * MB);
    _Float16* spk1 = (_Float16*)(ws + 14 * MB);

    // 1) Encoder: 512 blocks x 256 threads, 8 features/thread.
    snn_encode<<<SNN_B, 256, 0, stream>>>(x, cnt);

    // 2) Weight hi/lo splits.
    const int n1 = SNN_H1 * SNN_F;   // 2M
    const int n2 = SNN_H2 * SNN_H1;  // 1M
    snn_split_w<<<(n1 + 255) / 256, 256, 0, stream>>>(W1, w1hi, w1lo, n1);
    snn_split_w<<<(n2 + 255) / 256, 256, 0, stream>>>(W2, w2hi, w2lo, n2);

    // 3) GEMM1 + LIF1: 512 wave-tiles (4 M-tiles each), 8 waves per block.
    snn_gemm1_lif1<<<64, 256, 0, stream>>>(cnt, w1hi, w1lo, b1, spk1);

    // 4) GEMM2 (16 steps) + LIF2: 32 mt x 8 n-groups = 256 blocks.
    snn_gemm2_lif2<<<256, 256, 0, stream>>>(spk1, w2hi, w2lo, b2, out);
}
